// FactorAtt_ConvRelPosEnc_SK_37142877176322
// MI455X (gfx1250) — compile-verified, matrix operand order checked by reading, not run on hardware
//
#include <hip/hip_runtime.h>
#include <hip/hip_bf16.h>

// ---------------------------------------------------------------------------
// CDNA5 / gfx1250 wave32 WMMA implementation of CoaT FactorAtt+CRPE+SK block.
// bf16 inputs to v_wmma_f32_16x16x32_bf16, fp32 accumulate & epilogues.
// LDS tiles are stored in *fragment-linear* order (one contiguous 32B run per
// lane per 16x32 tile): fragment loads are 2x ds_load_b128; staging is
// float4 global loads -> vector fptrunc (v_cvt_pk_bf16_f32) -> ds_store_b64.
// ---------------------------------------------------------------------------

typedef __attribute__((ext_vector_type(16))) __bf16 bf16x16;
typedef __attribute__((ext_vector_type(4)))  __bf16 bf16x4;
typedef __attribute__((ext_vector_type(2)))  __bf16 bf16x2;
typedef __attribute__((ext_vector_type(8)))  float  f32x8;
typedef __attribute__((ext_vector_type(4)))  float  f32x4;
typedef __attribute__((ext_vector_type(2)))  float  f32x2;

#define BDIM   8
#define NDIM   3136
#define CDIM   512
#define HHEADS 8
#define CHDIM  64
#define QKVC   1536

__device__ __forceinline__ f32x8 wmma_bf16(bf16x16 a, bf16x16 b, f32x8 c) {
  return __builtin_amdgcn_wmma_f32_16x16x32_bf16(false, a, false, b,
                                                 (short)0, c, false, false);
}

// Inverse of the WMMA 16-bit A/B fragment mapping (ISA 7.12.2):
// element (r, c) of a 16x32 tile lives at lane=(r&15)|((c>>3)&1)<<4,
// in-lane slot e=((c>>4)<<3)|(c&7).  For c%4==0, (c..c+3) occupy the same
// lane at two adjacent u32 words (uint2-aligned).
__device__ __forceinline__ int frag_word(int r, int c2) {   // u32-word index
  const int lane = (r & 15) | (((c2 >> 3) & 1) << 4);
  const int e    = ((c2 >> 4) << 3) | (c2 & 7);
  return ((r >> 4) * 512 + lane * 16 + e) >> 1;
}

// Vector fptrunc: lowers to naturally-paired v_cvt_pk_bf16_f32.
__device__ __forceinline__ unsigned pack_bf16(float a, float b) {
  f32x2 f; f.x = a; f.y = b;
  bf16x2 h = __builtin_convertvector(f, bf16x2);
  unsigned u; __builtin_memcpy(&u, &h, 4);
  return u;
}

__device__ __forceinline__ uint2 pack4_bf16(float4 v) {
  f32x4 f; f.x = v.x; f.y = v.y; f.z = v.z; f.w = v.w;
  bf16x4 h = __builtin_convertvector(f, bf16x4);
  uint2 u; __builtin_memcpy(&u, &h, 8);
  return u;
}

// One contiguous 32-byte fragment load per lane.
__device__ __forceinline__ bf16x16 frag_ld(const unsigned* base, int tile, int lane) {
  return *(const bf16x16*)((const __bf16*)base + tile * 512 + lane * 16);
}

// ---------------------------------------------------------------------------
// 0) zero small accumulator buffer (deterministic per launch)
__global__ void zero_f32_kernel(float* p, int n) {
  int i = blockIdx.x * blockDim.x + threadIdx.x;
  if (i < n) p[i] = 0.f;
}

// ---------------------------------------------------------------------------
// 1) qkv = x @ w_qkv^T      M=25088, N=1536, K=512, BM=BN=128, KB=32
__global__ __launch_bounds__(256) void qkv_gemm_kernel(
    const float* __restrict__ x, const float* __restrict__ w,
    float* __restrict__ qkv) {
  const int K  = CDIM;
  const int n0 = blockIdx.x * 128;
  const int m0 = blockIdx.y * 128;
  __shared__ __attribute__((aligned(32))) unsigned Asm[2048];  // 8 tiles
  __shared__ __attribute__((aligned(32))) unsigned Bsm[2048];  // 8 tiles
  const int tid = threadIdx.x, lane = tid & 31, wid = tid >> 5;
  const int wr = wid >> 1, wc = wid & 1;   // wave: 32 rows x 64 cols
  const int ra = tid >> 3;                 // 0..31
  const int c4 = (tid & 7) * 4;            // k, multiple of 4
  const int dw = frag_word(ra, c4) >> 1;   // uint2 index; +256 per 32 rows
  f32x8 acc[2][4] = {};
  for (int kc = 0; kc < K; kc += 32) {
    if (kc) __syncthreads();
    const float* pA = x + (size_t)(m0 + ra) * K + kc + c4;
    const float* pB = w + (size_t)(n0 + ra) * K + kc + c4;
    if (kc + 32 < K) {
      __builtin_prefetch(pA + 32, 0, 1);
      __builtin_prefetch(pB + 32, 0, 1);
    }
#pragma unroll
    for (int j = 0; j < 4; ++j) {          // 32 rows per step
      const float4 va = *(const float4*)(pA + (size_t)j * 32 * K);
      const float4 vb = *(const float4*)(pB + (size_t)j * 32 * K);
      ((uint2*)Asm)[dw + j * 256] = pack4_bf16(va);
      ((uint2*)Bsm)[dw + j * 256] = pack4_bf16(vb);
    }
    __syncthreads();
    bf16x16 af[2], bf[4];
#pragma unroll
    for (int i = 0; i < 2; ++i) af[i] = frag_ld(Asm, wr * 2 + i, lane);
#pragma unroll
    for (int i = 0; i < 4; ++i) bf[i] = frag_ld(Bsm, wc * 4 + i, lane);
#pragma unroll
    for (int i = 0; i < 2; ++i)
#pragma unroll
      for (int j2 = 0; j2 < 4; ++j2)
        acc[i][j2] = wmma_bf16(af[i], bf[j2], acc[i][j2]);
  }
#pragma unroll
  for (int i = 0; i < 2; ++i)
#pragma unroll
    for (int j2 = 0; j2 < 4; ++j2)
#pragma unroll
      for (int r = 0; r < 8; ++r) {
        const int row = m0 + wr * 32 + i * 16 + r + ((lane >> 4) << 3);
        const int col = n0 + wc * 64 + j2 * 16 + (lane & 15);
        qkv[(size_t)row * QKVC + col] = acc[i][j2][r];
      }
}

// ---------------------------------------------------------------------------
// 2) per-(b,h,ch) online softmax stats over N for k
__global__ __launch_bounds__(256) void kstats_kernel(
    const float* __restrict__ qkv, float* __restrict__ kmax,
    float* __restrict__ ksum) {
  const int bh = blockIdx.x;                  // 64
  const int b = bh >> 3, h = bh & 7;
  const int ch = threadIdx.x & 63, g = threadIdx.x >> 6;   // 4 row groups
  const float* kb = qkv + (size_t)b * NDIM * QKVC + CDIM + h * CHDIM + ch;
  float m = -3.4e38f, s = 0.f;
  for (int n = g; n < NDIM; n += 4) {
    const float v = kb[(size_t)n * QKVC];
    if (v > m) { s = s * __expf(m - v) + 1.f; m = v; }
    else       { s += __expf(v - m); }
  }
  __shared__ float mm[256], ss[256];
  mm[threadIdx.x] = m; ss[threadIdx.x] = s;
  __syncthreads();
  if (g == 0) {
    for (int j = 1; j < 4; ++j) {
      const float m2 = mm[j * 64 + ch], s2 = ss[j * 64 + ch];
      if (m2 > m) { s = s * __expf(m - m2) + s2; m = m2; }
      else        { s += s2 * __expf(m2 - m); }
    }
    kmax[bh * 64 + ch] = m;
    ksum[bh * 64 + ch] = s;
  }
}

// ---------------------------------------------------------------------------
// 3) kv[ck][cv] = sum_n exp(k-max)^T v / sum   (64x64 per (b,h), K-loop over N)
__global__ __launch_bounds__(256) void kv_gemm_kernel(
    const float* __restrict__ qkv, const float* __restrict__ kmax,
    const float* __restrict__ ksum, float* __restrict__ kv) {
  const int bh = blockIdx.x;
  const int b = bh >> 3, h = bh & 7;
  __shared__ __attribute__((aligned(32))) unsigned Asm[1024];  // 4 tiles (64xK32)
  __shared__ __attribute__((aligned(32))) unsigned Bsm[1024];  // 4 tiles (K32x64)
  __shared__ float kmx[64];
  const int tid = threadIdx.x, lane = tid & 31, wid = tid >> 5;
  const int wr = wid >> 1, wc = wid & 1;      // wave: 16 rows x 32 cols
  if (tid < 64) kmx[tid] = kmax[bh * 64 + tid];
  f32x8 acc[2] = {};
  const float* kb = qkv + (size_t)b * NDIM * QKVC + CDIM + h * CHDIM;
  const float* vb = qkv + (size_t)b * NDIM * QKVC + 2 * CDIM + h * CHDIM;
  const int ch  = tid & 63;             // channel (A row / B col)
  const int nkb = (tid >> 6) * 2;       // even n-pair base, + j*8
  for (int n0 = 0; n0 < NDIM; n0 += 32) {
    __syncthreads();
    const float* pk = kb + (size_t)(n0 + nkb) * QKVC + ch;
    const float* pv = vb + (size_t)(n0 + nkb) * QKVC + ch;
    const float mx = kmx[ch];
#pragma unroll
    for (int j = 0; j < 4; ++j) {
      const int nk = nkb + j * 8;
      const float k0 = pk[(size_t)j * 8 * QKVC];
      const float k1 = pk[(size_t)j * 8 * QKVC + QKVC];
      const float v0 = pv[(size_t)j * 8 * QKVC];
      const float v1 = pv[(size_t)j * 8 * QKVC + QKVC];
      const int d = frag_word(ch, nk);        // pair along K(=n)
      Asm[d] = pack_bf16(__expf(k0 - mx), __expf(k1 - mx));
      Bsm[d] = pack_bf16(v0, v1);
    }
    __syncthreads();
    const bf16x16 fa = frag_ld(Asm, wr, lane);
    const bf16x16 b0 = frag_ld(Bsm, wc * 2, lane);
    const bf16x16 b1 = frag_ld(Bsm, wc * 2 + 1, lane);
    acc[0] = wmma_bf16(fa, b0, acc[0]);
    acc[1] = wmma_bf16(fa, b1, acc[1]);
  }
#pragma unroll
  for (int t = 0; t < 2; ++t)
#pragma unroll
    for (int r = 0; r < 8; ++r) {
      const int ck = wr * 16 + r + ((lane >> 4) << 3);
      const int cv = wc * 32 + t * 16 + (lane & 15);
      kv[(size_t)bh * 4096 + ck * 64 + cv] = acc[t][r] / ksum[bh * 64 + ck];
    }
}

// ---------------------------------------------------------------------------
// 4) depthwise conv of v: window 3 (heads 0-1), 5 (2-4), 7 (5-7)
__global__ __launch_bounds__(256) void dwconv_kernel(
    const float* __restrict__ qkv, const float* __restrict__ w3,
    const float* __restrict__ w5, const float* __restrict__ w7,
    float* __restrict__ convv) {
  const int wg = blockIdx.x;                  // b*448 + head*56 + y
  const int y = wg % 56; const int t = wg / 56;
  const int head = t % 8, b = t / 8;
  const int ksz = (head < 2) ? 3 : ((head < 5) ? 5 : 7);
  const int kk = ksz * ksz;
  __shared__ float wsm[64 * 49];
  const float* wsrc; int cbase;
  if (head < 2)      { wsrc = w3; cbase = head * 64; }
  else if (head < 5) { wsrc = w5; cbase = head * 64 - 128; }
  else               { wsrc = w7; cbase = head * 64 - 320; }
  for (int i = threadIdx.x; i < 64 * kk; i += blockDim.x)
    wsm[i] = wsrc[(size_t)(cbase + i / kk) * kk + (i % kk)];
  __syncthreads();
  const int pad = ksz >> 1;
  const float* vb = qkv + (size_t)b * NDIM * QKVC + 2 * CDIM + head * CHDIM;
  for (int li = threadIdx.x; li < 56 * 32; li += blockDim.x) {
    const int x = li >> 5, ch = (li & 31) * 2;    // two channels per thread
    float a0 = 0.f, a1 = 0.f;
    for (int dy = 0; dy < ksz; ++dy) {
      const int yy = y + dy - pad;
      if (yy < 0 || yy >= 56) continue;
      for (int dx = 0; dx < ksz; ++dx) {
        const int xx = x + dx - pad;
        if (xx < 0 || xx >= 56) continue;
        const float2 v = *(const float2*)(vb + (size_t)(yy * 56 + xx) * QKVC + ch);
        a0 += v.x * wsm[ch * kk + dy * ksz + dx];
        a1 += v.y * wsm[(ch + 1) * kk + dy * ksz + dx];
      }
    }
    float2 o; o.x = a0; o.y = a1;
    *(float2*)(convv + (((size_t)(b * 8 + head)) * NDIM + (y * 56 + x)) * 64 + ch) = o;
  }
}

// ---------------------------------------------------------------------------
// 5) fa = 0.125*(q@kv) + q*conv_v  + partial head/N reduction into da
__global__ __launch_bounds__(256) void fa_fused_kernel(
    const float* __restrict__ qkv, const float* __restrict__ kv,
    const float* __restrict__ convv, float* __restrict__ fa,
    float* __restrict__ da) {
  const int bh = blockIdx.y;
  const int b = bh >> 3, h = bh & 7;
  const int m0 = blockIdx.x * 64;
  __shared__ __attribute__((aligned(32))) unsigned Qsm[2][1024];  // K-slab, 4 tiles
  __shared__ __attribute__((aligned(32))) unsigned KVsm[2][1024];
  __shared__ float red[64 * 64];
  const int tid = threadIdx.x, lane = tid & 31, wid = tid >> 5;
  const int wr = wid & 1, wc = wid >> 1;    // wave: 32 rows x 16 cols
  const float* qb = qkv + (size_t)b * NDIM * QKVC + h * CHDIM;
  const float* kvb = kv + (size_t)bh * 4096;
  // stage q tile [64 x 64]: float4 loads, one ds_store_b64 per 4 elements
  {
    const int r0 = tid >> 4;                // 0..15, + j*16
    const int qc4 = (tid & 15) * 4;         // 0..60
    const int slab = qc4 >> 5;
    const int dw = frag_word(r0, qc4 & 31) >> 1;   // +128 uint2 per 16 rows
    const float* pq = qb + (size_t)(m0 + r0) * QKVC + qc4;
#pragma unroll
    for (int j = 0; j < 4; ++j) {
      const float4 v = *(const float4*)(pq + (size_t)j * 16 * QKVC);
      ((uint2*)Qsm[slab])[dw + j * 128] = pack4_bf16(v);
    }
  }
  // stage kv [64 x 64]: pair along k(=ck), strided loads, coalesced over cv
  {
    const int cv = tid & 63;
    const int ckb = (tid >> 6) * 2;         // + j*8
#pragma unroll
    for (int j = 0; j < 8; ++j) {
      const int ck = ckb + j * 8;
      const float u0 = kvb[(size_t)ck * 64 + cv];
      const float u1 = kvb[(size_t)(ck + 1) * 64 + cv];
      KVsm[ck >> 5][frag_word(cv, ck & 31)] = pack_bf16(u0, u1);
    }
  }
  __syncthreads();
  f32x8 acc[2] = {};
#pragma unroll
  for (int s = 0; s < 2; ++s) {
    const bf16x16 a0 = frag_ld(Qsm[s], wr * 2, lane);
    const bf16x16 a1 = frag_ld(Qsm[s], wr * 2 + 1, lane);
    const bf16x16 bf = frag_ld(KVsm[s], wc, lane);
    acc[0] = wmma_bf16(a0, bf, acc[0]);
    acc[1] = wmma_bf16(a1, bf, acc[1]);
  }
  const float scale = 0.125f;   // 64^-0.5
#pragma unroll
  for (int t = 0; t < 2; ++t)
#pragma unroll
    for (int r = 0; r < 8; ++r) {
      const int row = wr * 32 + t * 16 + r + ((lane >> 4) << 3);
      const int col = wc * 16 + (lane & 15);
      const size_t g = ((size_t)bh * NDIM + m0 + row) * 64 + col;
      const float qv = qb[(size_t)(m0 + row) * QKVC + col];
      const float v = scale * acc[t][r] + qv * convv[g];
      fa[g] = v;
      red[row * 64 + col] = v;
    }
  __syncthreads();
  if (tid < 64) {
    float s = 0.f;
    for (int r = 0; r < 64; ++r) s += red[r * 64 + tid];
    atomicAdd(&da[b * 64 + tid], s);
  }
}

// ---------------------------------------------------------------------------
// 6) SK gate: mean over N -> MLP+BN+ReLU -> sel -> softmax over heads
__global__ __launch_bounds__(256) void sk_kernel(
    const float* __restrict__ da, const float* __restrict__ w_tr,
    const float* __restrict__ gam, const float* __restrict__ bet,
    const float* __restrict__ mu, const float* __restrict__ var,
    const float* __restrict__ w_sel, float* __restrict__ sel) {
  __shared__ float tsm[8 * 32];
  __shared__ float ssm[8 * 512];
  const int tid = threadIdx.x;
  {
    const int b = tid >> 5, j = tid & 31;
    float acc = 0.f;
    for (int ck = 0; ck < 64; ++ck)
      acc += (da[b * 64 + ck] * (1.f / 3136.f)) * w_tr[j * 64 + ck];
    acc = (acc - mu[j]) * gam[j] * rsqrtf(var[j] + 1e-5f) + bet[j];
    tsm[b * 32 + j] = acc > 0.f ? acc : 0.f;
  }
  __syncthreads();
  for (int i = tid; i < 4096; i += 256) {
    const int b = i >> 9, hc = i & 511;
    float acc = 0.f;
    for (int j = 0; j < 32; ++j) acc += tsm[b * 32 + j] * w_sel[hc * 32 + j];
    ssm[i] = acc;
  }
  __syncthreads();
  for (int p = tid; p < 512; p += 256) {
    const int b = p >> 6, ck = p & 63;
    float m = -3.4e38f;
    for (int hh = 0; hh < 8; ++hh) m = fmaxf(m, ssm[b * 512 + hh * 64 + ck]);
    float e[8], s = 0.f;
    for (int hh = 0; hh < 8; ++hh) { e[hh] = __expf(ssm[b * 512 + hh * 64 + ck] - m); s += e[hh]; }
    for (int hh = 0; hh < 8; ++hh) sel[b * 512 + hh * 64 + ck] = e[hh] / s;
  }
}

// ---------------------------------------------------------------------------
// 7) out = (sel * fa) @ w_proj^T + b_proj   per-b GEMM, gating fused in A-load
__global__ __launch_bounds__(256) void out_gemm_kernel(
    const float* __restrict__ fa, const float* __restrict__ sel,
    const float* __restrict__ w, const float* __restrict__ bias,
    float* __restrict__ out) {
  const int K  = CDIM;
  const int n0 = blockIdx.x * 128;
  const int m0 = blockIdx.y * 64;
  const int b  = blockIdx.z;
  __shared__ __attribute__((aligned(32))) unsigned Asm[1024];  // 4 tiles (64xK32)
  __shared__ __attribute__((aligned(32))) unsigned Bsm[2048];  // 8 tiles (K32x128)
  const int tid = threadIdx.x, lane = tid & 31, wid = tid >> 5;
  const int wr = wid & 1, wc = wid >> 1;     // wave: 32 rows x 32 cols
  const int ra = tid >> 3;                   // 0..31
  const int c4 = (tid & 7) * 4;              // k, multiple of 4
  const int dw = frag_word(ra, c4) >> 1;     // uint2 index; +256 per 32 rows
  f32x8 acc[2][2] = {};
  for (int kc = 0; kc < K; kc += 32) {
    if (kc) __syncthreads();
    const int kcol = kc + c4;
    const int hh = kcol >> 6, ch = kcol & 63;     // ch..ch+3 same head
    const float4 sv = *(const float4*)(sel + b * 512 + kcol);
    const float* pA = fa + (((size_t)(b * 8 + hh)) * NDIM + m0 + ra) * 64 + ch;
    const float* pB = w + (size_t)(n0 + ra) * K + kc + c4;
    if (kc + 32 < K)
      __builtin_prefetch(pB + 32, 0, 1);
#pragma unroll
    for (int j = 0; j < 2; ++j) {               // A: 64 rows
      float4 fv = *(const float4*)(pA + (size_t)j * 32 * 64);
      fv.x *= sv.x; fv.y *= sv.y; fv.z *= sv.z; fv.w *= sv.w;
      ((uint2*)Asm)[dw + j * 256] = pack4_bf16(fv);
    }
#pragma unroll
    for (int j = 0; j < 4; ++j) {               // B: 128 cols
      const float4 wv = *(const float4*)(pB + (size_t)j * 32 * K);
      ((uint2*)Bsm)[dw + j * 256] = pack4_bf16(wv);
    }
    __syncthreads();
    bf16x16 af[2], bf[2];
#pragma unroll
    for (int i = 0; i < 2; ++i) af[i] = frag_ld(Asm, wr * 2 + i, lane);
#pragma unroll
    for (int i = 0; i < 2; ++i) bf[i] = frag_ld(Bsm, wc * 2 + i, lane);
#pragma unroll
    for (int i = 0; i < 2; ++i)
#pragma unroll
      for (int j2 = 0; j2 < 2; ++j2)
        acc[i][j2] = wmma_bf16(af[i], bf[j2], acc[i][j2]);
  }
#pragma unroll
  for (int i = 0; i < 2; ++i)
#pragma unroll
    for (int j2 = 0; j2 < 2; ++j2)
#pragma unroll
      for (int r = 0; r < 8; ++r) {
        const int row = m0 + wr * 32 + i * 16 + r + ((lane >> 4) << 3);
        const int col = n0 + wc * 32 + j2 * 16 + (lane & 15);
        out[((size_t)b * NDIM + row) * CDIM + col] = acc[i][j2][r] + bias[col];
      }
}

// ---------------------------------------------------------------------------
extern "C" void kernel_launch(void* const* d_in, const int* in_sizes, int n_in,
                              void* d_out, int out_size, void* d_ws, size_t ws_size,
                              hipStream_t stream) {
  (void)in_sizes; (void)n_in; (void)out_size; (void)ws_size;
  const float* x      = (const float*)d_in[0];
  const float* w_qkv  = (const float*)d_in[1];
  const float* w_proj = (const float*)d_in[2];
  const float* b_proj = (const float*)d_in[3];
  const float* w_tr   = (const float*)d_in[4];
  const float* bn_g   = (const float*)d_in[5];
  const float* bn_b   = (const float*)d_in[6];
  const float* bn_m   = (const float*)d_in[7];
  const float* bn_v   = (const float*)d_in[8];
  const float* w_sel  = (const float*)d_in[9];
  const float* w3     = (const float*)d_in[10];
  const float* w5     = (const float*)d_in[11];
  const float* w7     = (const float*)d_in[12];
  float* out = (float*)d_out;

  // workspace layout (floats)
  float* ws   = (float*)d_ws;
  float* qkvb = ws;  ws += (size_t)BDIM * NDIM * QKVC;           // 38.5M
  float* convv = ws; ws += (size_t)BDIM * HHEADS * NDIM * CHDIM; // 12.8M
  float* fab  = ws;  ws += (size_t)BDIM * HHEADS * NDIM * CHDIM; // 12.8M
  float* kvb  = ws;  ws += (size_t)64 * 64 * 64;                 // 262K
  float* kmaxb = ws; ws += 4096;
  float* ksumb = ws; ws += 4096;
  float* dab  = ws;  ws += 512;
  float* selb = ws;  ws += 4096;

  zero_f32_kernel<<<2, 256, 0, stream>>>(dab, 512);
  qkv_gemm_kernel<<<dim3(12, 196), 256, 0, stream>>>(x, w_qkv, qkvb);
  kstats_kernel<<<64, 256, 0, stream>>>(qkvb, kmaxb, ksumb);
  kv_gemm_kernel<<<64, 256, 0, stream>>>(qkvb, kmaxb, ksumb, kvb);
  dwconv_kernel<<<BDIM * HHEADS * 56, 256, 0, stream>>>(qkvb, w3, w5, w7, convv);
  fa_fused_kernel<<<dim3(49, 64), 256, 0, stream>>>(qkvb, kvb, convv, fab, dab);
  sk_kernel<<<1, 256, 0, stream>>>(dab, w_tr, bn_g, bn_b, bn_m, bn_v, w_sel, selb);
  out_gemm_kernel<<<dim3(4, 49, 8), 256, 0, stream>>>(fab, selb, w_proj, b_proj, out);
}
